// MultiScaleAggregator_28097676050879
// MI455X (gfx1250) — matrix operand order repaired and checked by reading, not compiled, not run on hardware
//
#include <hip/hip_runtime.h>
#include <math.h>

// Problem constants
#define BB 1024
#define NN 200
#define EE 256
#define RR 256

typedef __attribute__((ext_vector_type(16))) __bf16 v16bf;
typedef __attribute__((ext_vector_type(8)))  __bf16 bf16x8;
typedef __attribute__((ext_vector_type(8)))  float  v8f;

// GCC-style int4 vector with explicit address spaces, matching the
// __builtin_amdgcn_global_load_async_to_lds_b128 signature:
//   (int4 __device__* src, int4 __shared__* dst, imm offset, imm cpol)
typedef int v4i_t __attribute__((vector_size(16)));
typedef __attribute__((address_space(1))) v4i_t as1_v4i;
typedef __attribute__((address_space(3))) v4i_t as3_v4i;

#if defined(__HIP_DEVICE_COMPILE__) && __has_builtin(__builtin_amdgcn_global_load_async_to_lds_b128)
#define HAVE_ASYNC_LDS 1
#else
#define HAVE_ASYNC_LDS 0
#endif

__device__ __forceinline__ void async_wait0() {
#if __has_builtin(__builtin_amdgcn_s_wait_asynccnt)
  __builtin_amdgcn_s_wait_asynccnt(0);
#else
  asm volatile("s_wait_asynccnt 0" ::: "memory");
#endif
}

// ---------------------------------------------------------------------------
// WMMA helpers (CDNA5 16x16x32 bf16, fp32 accumulate)
// ---------------------------------------------------------------------------
__device__ __forceinline__ v8f wmma_bf16(v16bf a, v16bf b, v8f c) {
  // 8 args: (neg_a, A, neg_b, B, c_mod, C, reuse_a, reuse_b)
  return __builtin_amdgcn_wmma_f32_16x16x32_bf16(false, a, false, b, (short)0, c,
                                                 false, false);
}

// Build a 16-element bf16 fragment from two 8-element (16B) chunks.
__device__ __forceinline__ v16bf frag16(const __bf16* lo, const __bf16* hi) {
  bf16x8 a = *reinterpret_cast<const bf16x8*>(lo);
  bf16x8 b = *reinterpret_cast<const bf16x8*>(hi);
  v16bf r;
#pragma unroll
  for (int j = 0; j < 8; ++j) { r[j] = a[j]; r[j + 8] = b[j]; }
  return r;
}

// A-matrix 16x32 bf16 fragment from an LDS row.
// lanes 0-15: K chunks {0..7},{16..23}; lanes 16-31: {8..15},{24..31}
__device__ __forceinline__ v16bf load_a_frag(const __bf16* row_ptr, int kt, int lane) {
  int base = (lane & 16) ? 8 : 0;
  const __bf16* p = row_ptr + kt * 32 + base;
  return frag16(p, p + 16);
}

// B-matrix 32x16 bf16 fragment from packed memory: P[kt][col][kk], kk in [0,32).
// Lane col = lane&15; lanes 0-15: K chunks {0..7},{16..23}; lanes 16-31: {8..15},{24..31}
__device__ __forceinline__ v16bf load_b_frag(const __bf16* packed_col, int lane) {
  int kbase = (lane & 16) ? 8 : 0;
  const __bf16* p = packed_col + kbase;
  return frag16(p, p + 16);
}

// ---------------------------------------------------------------------------
// K0: pack rel_w [E][R] f32 -> bf16 P[kt=8][col=E][kk=32]  (B-matrix layout)
// ---------------------------------------------------------------------------
__global__ __launch_bounds__(256) void k_pack_relw(const float* __restrict__ rel_w,
                                                   unsigned short* __restrict__ dst_u16) {
  __bf16* dst = reinterpret_cast<__bf16*>(dst_u16);
  int i = blockIdx.x * 256 + threadIdx.x;            // i over E*R = 65536
  int e = i >> 8;
  int k = i & 255;
  int kt = k >> 5, kk = k & 31;
  dst[((size_t)kt * 256 + e) * 32 + kk] = (__bf16)rel_w[(size_t)e * 256 + k];
}

// ---------------------------------------------------------------------------
// K0b: pack sp_w [3][E][6E] f32 -> bf16 P[s][kt=48][col=E][kk=32]
// ---------------------------------------------------------------------------
__global__ __launch_bounds__(256) void k_pack_spw(const float* __restrict__ sp_w,
                                                  unsigned short* __restrict__ dst_u16) {
  __bf16* dst = reinterpret_cast<__bf16*>(dst_u16);
  size_t i = (size_t)blockIdx.x * 256 + threadIdx.x;  // over 3*256*1536
  int s = (int)(i / (256 * 1536));
  int rem = (int)(i % (256 * 1536));
  int j = rem / 1536;
  int k = rem % 1536;
  int kt = k >> 5, kk = k & 31;
  dst[(((size_t)s * 48 + kt) * 256 + j) * 32 + kk] =
      (__bf16)sp_w[((size_t)s * 256 + j) * 1536 + k];
}

// ---------------------------------------------------------------------------
// K1: q_key[b,e] = dot(r_emb[b,:], rel_w[e,:])
// ---------------------------------------------------------------------------
__global__ __launch_bounds__(256) void k_qkey(const float* __restrict__ r_emb,
                                              const float* __restrict__ rel_w,
                                              float* __restrict__ qkey) {
  __shared__ float sR[256];
  int b = blockIdx.x, e = threadIdx.x;
  sR[e] = r_emb[(size_t)b * 256 + e];
  __syncthreads();
  const float* w = rel_w + (size_t)e * 256;
  float acc = 0.f;
#pragma unroll 4
  for (int k = 0; k < 256; k += 4) {
    float4 v = *reinterpret_cast<const float4*>(w + k);
    acc += v.x * sR[k] + v.y * sR[k + 1] + v.z * sR[k + 2] + v.w * sR[k + 3];
  }
  qkey[(size_t)b * 256 + e] = acc;
}

// ---------------------------------------------------------------------------
// K2: q_proj[s,b,j] = dot(q_key[b,:], attn_w[s][j,:])
// ---------------------------------------------------------------------------
__global__ __launch_bounds__(256) void k_qproj(const float* __restrict__ qkey,
                                               const float* __restrict__ attn_w,
                                               float* __restrict__ qproj) {
  __shared__ float sQ[256];
  int b = blockIdx.x, s = blockIdx.y, j = threadIdx.x;
  sQ[j] = qkey[(size_t)b * 256 + j];
  __syncthreads();
  const float* w = attn_w + ((size_t)s * 256 + j) * 256;
  float acc = 0.f;
#pragma unroll 4
  for (int k = 0; k < 256; k += 4) {
    float4 v = *reinterpret_cast<const float4*>(w + k);
    acc += v.x * sQ[k] + v.y * sQ[k + 1] + v.z * sQ[k + 2] + v.w * sQ[k + 3];
  }
  qproj[((size_t)s * BB + b) * 256 + j] = acc;
}

// ---------------------------------------------------------------------------
// K3: cross-scale softmax weights[b][3]
// ---------------------------------------------------------------------------
__global__ __launch_bounds__(256) void k_csw(const float* __restrict__ r_emb,
                                             const float* __restrict__ csq_w,
                                             const float* __restrict__ csq_b,
                                             const float* __restrict__ csl_w,
                                             const float* __restrict__ csl_b,
                                             float* __restrict__ wts) {
  __shared__ float sR[256];
  __shared__ float sSq[256];
  __shared__ float sL[3];
  int b = blockIdx.x, e = threadIdx.x;
  sR[e] = r_emb[(size_t)b * 256 + e];
  __syncthreads();
  const float* w = csq_w + (size_t)e * 256;
  float acc = csq_b[e];
  for (int k = 0; k < 256; ++k) acc += w[k] * sR[k];
  sSq[e] = acc;
  __syncthreads();
  if (e < 3) {
    float l = csl_b[e];
    const float* cw = csl_w + (size_t)e * 256;
    for (int k = 0; k < 256; ++k) l += cw[k] * sSq[k];
    sL[e] = l;
  }
  __syncthreads();
  if (e == 0) {
    float m = fmaxf(sL[0], fmaxf(sL[1], sL[2]));
    float e0 = expf(sL[0] - m), e1 = expf(sL[1] - m), e2 = expf(sL[2] - m);
    float inv = 1.f / (e0 + e1 + e2);
    wts[(size_t)b * 3 + 0] = e0 * inv;
    wts[(size_t)b * 3 + 1] = e1 * inv;
    wts[(size_t)b * 3 + 2] = e2 * inv;
  }
}

// ---------------------------------------------------------------------------
// K4: MAIN fused kernel. One workgroup (8 wave32) per batch row b.
// WMMA bf16 GEMM msgs[b] = nb_rel[b] @ rel_w^T  tile-by-tile (32 rows x 256),
// fused masked PNA stats + flash-style online-softmax attention.
// Staging uses GLOBAL_LOAD_ASYNC_TO_LDS (ASYNCcnt) when available.
// ---------------------------------------------------------------------------
__global__ __launch_bounds__(256) void k_main(
    const float* __restrict__ e_emb, const float* __restrict__ nb_rel,
    const float* __restrict__ delta_t, const int* __restrict__ hist_mask,
    const float* __restrict__ lt1, const float* __restrict__ lt2,
    const float* __restrict__ shp, const float* __restrict__ lg,
    const unsigned short* __restrict__ relw_u16, const float* __restrict__ qproj,
    unsigned short* __restrict__ feats_u16) {
  const __bf16* relw_pk = reinterpret_cast<const __bf16*>(relw_u16);
  __bf16* feats = reinterpret_cast<__bf16*>(feats_u16);

  __shared__ __bf16 sA[32 * 256];     // bf16 A tile (32 rows x 256 K)     16 KB
  __shared__ float  sM[32 * 256];     // fp32 msgs tile / async landing    32 KB
  __shared__ float  sQP[3 * 256];     // q_proj for this b                  3 KB
  __shared__ float  sCoef[3 * 224];   // decay*scale_w per (s,n)
  __shared__ float  sMaskf[224];
  __shared__ float  sAP[32 * 8 * 3];  // attn dot partials
  __shared__ float  sRaw[3 * 32];
  __shared__ float  sQexp[3 * 32];
  __shared__ float  sResc[3];
  __shared__ float  sRunMax[3];
  __shared__ float  sRunDen[3];
  __shared__ float  sNV, sAllInv;

  const int b = blockIdx.x;
  const int tid = threadIdx.x;
  const int lane = tid & 31;
  const int wave = tid >> 5;

  // scalars (broadcast loads)
  const float gamma = expf(lg[0]);
  const float tau1 = expf(lt1[0]);
  const float tau2 = expf(lt2[0]) + tau1;
  const float sharp = fminf(fmaxf(shp[0], 0.1f), 5.0f);

  // per-neighbor temporal coefficients
  if (tid < 224) {
    if (tid < NN) {
      float dt = delta_t[(size_t)b * NN + tid];
      int m = hist_mask[(size_t)b * NN + tid];
      float dec = expf(-gamma * fmaxf(dt, 0.f));
      float wf = 1.f / (1.f + expf(sharp * (dt - tau1)));   // sigmoid(-sharp*(dt-tau1))
      float wc = 1.f / (1.f + expf(-sharp * (dt - tau2)));  // sigmoid( sharp*(dt-tau2))
      float wm = fmaxf(1.f - wf - wc, 0.f);
      sCoef[0 * 224 + tid] = dec * wf;
      sCoef[1 * 224 + tid] = dec * wm;
      sCoef[2 * 224 + tid] = dec * wc;
      sMaskf[tid] = m ? 1.f : 0.f;
    } else {
      sCoef[0 * 224 + tid] = 0.f; sCoef[1 * 224 + tid] = 0.f; sCoef[2 * 224 + tid] = 0.f;
      sMaskf[tid] = 0.f;
    }
  }
#pragma unroll
  for (int s = 0; s < 3; ++s) sQP[s * 256 + tid] = qproj[((size_t)s * BB + b) * 256 + tid];
  if (tid < 3) { sRunMax[tid] = -1e30f; sRunDen[tid] = 0.f; }
  __syncthreads();
  if (tid == 0) {
    float c = 0.f;
    for (int n = 0; n < NN; ++n) c += sMaskf[n];
    sNV = fmaxf(c, 1.f);
    sAllInv = (c == 0.f) ? 1.f : 0.f;
  }
  __syncthreads();

  // per-thread PNA state (thread owns column e = tid)
  float vsum[3], vmx[3], vmn[3], vsq[3], vacc[3];
#pragma unroll
  for (int s = 0; s < 3; ++s) {
    vsum[s] = 0.f; vmx[s] = -3e38f; vmn[s] = 1e4f; vsq[s] = 0.f; vacc[s] = 0.f;
  }

  const int msub = wave & 1;
  const int eblk = (wave >> 1) * 64;

  for (int nt = 0; nt < 7; ++nt) {
    const int n0 = nt * 32;
    const int vr = min(32, NN - n0);
    const int row = tid >> 3, seg = tid & 7;

#if HAVE_ASYNC_LDS
    // ---- async DMA: nb_rel fp32 tile -> sM (LDS), then convert to bf16 sA ----
    if (row < vr) {
      const float* src = nb_rel + (((size_t)b * NN + n0 + row) * 256 + seg * 32);
      float* dst = &sM[row * 256 + seg * 32];
#pragma unroll
      for (int j = 0; j < 8; ++j)
        __builtin_amdgcn_global_load_async_to_lds_b128((as1_v4i*)(src + j * 4),
                                                       (as3_v4i*)(dst + j * 4), 0, 0);
    }
    async_wait0();
    __syncthreads();
    {
      __bf16* dst = &sA[row * 256 + seg * 32];
      if (row < vr) {
        const float* srcl = &sM[row * 256 + seg * 32];
#pragma unroll
        for (int j = 0; j < 32; ++j) dst[j] = (__bf16)srcl[j];
        if (nt + 1 < 7 && (n0 + 32 + row) < NN)
          __builtin_prefetch(nb_rel + (((size_t)b * NN + n0 + 32 + row) * 256 + seg * 32), 0, 1);
      } else {
#pragma unroll
        for (int j = 0; j < 32; ++j) dst[j] = (__bf16)0.f;
      }
    }
#else
    // ---- synchronous staging: nb_rel fp32 -> bf16 LDS ------------------------
    {
      __bf16* dst = &sA[row * 256 + seg * 32];
      if (row < vr) {
        const float* src = nb_rel + (((size_t)b * NN + n0 + row) * 256 + seg * 32);
#pragma unroll
        for (int j = 0; j < 32; j += 4) {
          float4 v = *reinterpret_cast<const float4*>(src + j);
          dst[j + 0] = (__bf16)v.x; dst[j + 1] = (__bf16)v.y;
          dst[j + 2] = (__bf16)v.z; dst[j + 3] = (__bf16)v.w;
        }
        if (nt + 1 < 7 && (n0 + 32 + row) < NN)
          __builtin_prefetch(nb_rel + (((size_t)b * NN + n0 + 32 + row) * 256 + seg * 32), 0, 1);
      } else {
#pragma unroll
        for (int j = 0; j < 32; ++j) dst[j] = (__bf16)0.f;
      }
    }
#endif
    __syncthreads();

    // ---- WMMA: msgs tile = A(32x256) * relw^T(256x256) -----------------------
    v8f c[4];
#pragma unroll
    for (int sub = 0; sub < 4; ++sub)
#pragma unroll
      for (int i = 0; i < 8; ++i) c[sub][i] = 0.f;

    const __bf16* arow = &sA[(msub * 16 + (lane & 15)) * 256];
#pragma unroll
    for (int kt = 0; kt < 8; ++kt) {
      v16bf af = load_a_frag(arow, kt, lane);
#pragma unroll
      for (int sub = 0; sub < 4; ++sub) {
        int col = eblk + sub * 16 + (lane & 15);
        const __bf16* bp = relw_pk + ((size_t)(kt * 256 + col)) * 32;
        v16bf bfr = load_b_frag(bp, lane);
        c[sub] = wmma_bf16(af, bfr, c[sub]);
      }
    }
    // store D fragments to LDS msgs tile
    {
      int rbase = msub * 16 + ((lane & 16) ? 8 : 0);
      int cl = lane & 15;
#pragma unroll
      for (int sub = 0; sub < 4; ++sub) {
        int col = eblk + sub * 16 + cl;
#pragma unroll
        for (int i = 0; i < 8; ++i) sM[(rbase + i) * 256 + col] = c[sub][i];
      }
    }
    __syncthreads();

    // ---- attention raw: partial dots over 32-e segments ----------------------
    {
      float p0 = 0.f, p1 = 0.f, p2 = 0.f;
      if (row < vr) {
        const float* mrow = &sM[row * 256 + seg * 32];
#pragma unroll 8
        for (int j = 0; j < 32; ++j) {
          float m = mrow[j];
          p0 += m * sQP[0 * 256 + seg * 32 + j];
          p1 += m * sQP[1 * 256 + seg * 32 + j];
          p2 += m * sQP[2 * 256 + seg * 32 + j];
        }
      }
      sAP[(row * 8 + seg) * 3 + 0] = p0;
      sAP[(row * 8 + seg) * 3 + 1] = p1;
      sAP[(row * 8 + seg) * 3 + 2] = p2;
    }
    __syncthreads();
    if (tid < 96) {
      int s = tid >> 5, r = tid & 31;
      if (r < vr) {
        float d = 0.f;
#pragma unroll
        for (int sg = 0; sg < 8; ++sg) d += sAP[(r * 8 + sg) * 3 + s];
        int n = n0 + r;
        float raw = d * 0.0625f * sCoef[s * 224 + n];  // /sqrt(E) and *dec*w_s
        raw = (sMaskf[n] > 0.f) ? raw : -10000.f;
        if (sAllInv > 0.f) raw = 0.f;
        sRaw[s * 32 + r] = raw;
      } else {
        sRaw[s * 32 + r] = -3e38f;
      }
    }
    __syncthreads();
    // online softmax bookkeeping (tiny, serialized over 3 scales)
    if (tid < 3) {
      int s = tid;
      float tm = -3e38f;
      for (int r = 0; r < vr; ++r) tm = fmaxf(tm, sRaw[s * 32 + r]);
      float nm = fmaxf(sRunMax[s], tm);
      float resc = expf(sRunMax[s] - nm);
      float dsum = 0.f;
      for (int r = 0; r < vr; ++r) {
        float q = expf(sRaw[s * 32 + r] - nm);
        sQexp[s * 32 + r] = q;
        dsum += q;
      }
      for (int r = vr; r < 32; ++r) sQexp[s * 32 + r] = 0.f;
      sRunDen[s] = sRunDen[s] * resc + dsum;
      sRunMax[s] = nm;
      sResc[s] = resc;
    }
    __syncthreads();

    // ---- per-column accumulation of stats + attention ------------------------
    {
      int e = tid;
#pragma unroll
      for (int s = 0; s < 3; ++s) vacc[s] *= sResc[s];
      for (int r = 0; r < vr; ++r) {
        float m = sM[r * 256 + e];
        int n = n0 + r;
        float msk = sMaskf[n];
#pragma unroll
        for (int s = 0; s < 3; ++s) {
          float cs = sCoef[s * 224 + n];
          float ms = m * cs;        // m_scales element
          float mv = ms * msk;      // msgs_v element
          vsum[s] += mv;
          vmx[s] = fmaxf(vmx[s], mv);
          vmn[s] = fminf(vmn[s], (msk > 0.f) ? ms : 1e4f);
          float cl = fminf(fmaxf(mv, -10.f), 10.f);
          vsq[s] += cl * cl;
          vacc[s] += sQexp[s * 32 + r] * ms;
        }
      }
    }
    __syncthreads();
  }

  // ---- finalize feats = [mean|max|min|std|attn|e_emb], write bf16 ------------
  {
    int e = tid;
    float nv = sNV;
    float ee = e_emb[(size_t)b * 256 + e];
#pragma unroll
    for (int s = 0; s < 3; ++s) {
      float mean = vsum[s] / nv;
      float sqm = vsq[s] / nv;
      float stdv = sqrtf(fmaxf(sqm - mean * mean, 1e-6f));
      float mnc = fminf(fmaxf(vmn[s], -1e4f), 1e4f);
      float attn = vacc[s] / sRunDen[s];
      __bf16* fb = feats + ((size_t)s * BB + b) * 1536;
      fb[0 * 256 + e] = (__bf16)mean;
      fb[1 * 256 + e] = (__bf16)vmx[s];
      fb[2 * 256 + e] = (__bf16)mnc;
      fb[3 * 256 + e] = (__bf16)stdv;
      fb[4 * 256 + e] = (__bf16)attn;
      fb[5 * 256 + e] = (__bf16)ee;
    }
  }
}

// ---------------------------------------------------------------------------
// K5: h = feats @ sp_w^T + sp_b ; layernorm ; exact gelu.  WMMA bf16.
// grid (B/32, 3), 256 threads. Stages 128 K-columns per barrier.
// ---------------------------------------------------------------------------
__global__ __launch_bounds__(256) void k_sp(const unsigned short* __restrict__ feats_u16,
                                            const unsigned short* __restrict__ spw_u16,
                                            const float* __restrict__ sp_b,
                                            const float* __restrict__ ln_g,
                                            const float* __restrict__ ln_b,
                                            float* __restrict__ scales_out) {
  const __bf16* feats = reinterpret_cast<const __bf16*>(feats_u16);
  const __bf16* spw_pk = reinterpret_cast<const __bf16*>(spw_u16);

  __shared__ __bf16 sA[32 * 128];     // 8 KB
  __shared__ float  sH[32 * 256];     // 32 KB
  __shared__ float  sPart[32 * 8 * 2];
  __shared__ float  sMV[32 * 2];

  const int s = blockIdx.y;
  const int b0 = blockIdx.x * 32;
  const int tid = threadIdx.x;
  const int lane = tid & 31;
  const int wave = tid >> 5;
  const int msub = wave & 1;
  const int eblk = (wave >> 1) * 64;

  v8f c[4];
#pragma unroll
  for (int sub = 0; sub < 4; ++sub)
#pragma unroll
    for (int i = 0; i < 8; ++i) c[sub][i] = 0.f;

  for (int ktb = 0; ktb < 12; ++ktb) {
    // stage 32 rows x 128 K bf16
    {
      int row = tid >> 3, kk = (tid & 7) * 16;
      const __bf16* src = feats + (((size_t)s * BB + b0 + row) * 1536 + ktb * 128 + kk);
      __bf16* dst = &sA[row * 128 + kk];
#pragma unroll
      for (int j = 0; j < 16; ++j) dst[j] = src[j];
    }
    __syncthreads();
#pragma unroll
    for (int k2 = 0; k2 < 4; ++k2) {
      const int kt = ktb * 4 + k2;
      const __bf16* arow = &sA[(msub * 16 + (lane & 15)) * 128 + k2 * 32];
      int base = (lane & 16) ? 8 : 0;
      v16bf af = frag16(arow + base, arow + base + 16);
#pragma unroll
      for (int sub = 0; sub < 4; ++sub) {
        int col = eblk + sub * 16 + (lane & 15);
        const __bf16* bp = spw_pk + (((size_t)s * 48 + kt) * 256 + col) * 32;
        v16bf bfr = load_b_frag(bp, lane);
        c[sub] = wmma_bf16(af, bfr, c[sub]);
      }
    }
    __syncthreads();
  }

  // D -> LDS with bias
  {
    int rbase = msub * 16 + ((lane & 16) ? 8 : 0);
    int cl = lane & 15;
#pragma unroll
    for (int sub = 0; sub < 4; ++sub) {
      int col = eblk + sub * 16 + cl;
#pragma unroll
      for (int i = 0; i < 8; ++i)
        sH[(rbase + i) * 256 + col] = c[sub][i] + sp_b[(size_t)s * 256 + col];
    }
  }
  __syncthreads();

  // layernorm stats (per row over 256 cols)
  {
    int r = tid >> 3, seg = tid & 7;
    float a = 0.f, q = 0.f;
    const float* hrow = &sH[r * 256 + seg * 32];
#pragma unroll 8
    for (int j = 0; j < 32; ++j) { float v = hrow[j]; a += v; q += v * v; }
    sPart[(r * 8 + seg) * 2 + 0] = a;
    sPart[(r * 8 + seg) * 2 + 1] = q;
  }
  __syncthreads();
  if (tid < 32) {
    int r = tid;
    float a = 0.f, q = 0.f;
#pragma unroll
    for (int seg = 0; seg < 8; ++seg) {
      a += sPart[(r * 8 + seg) * 2 + 0];
      q += sPart[(r * 8 + seg) * 2 + 1];
    }
    float mu = a * (1.f / 256.f);
    float var = q * (1.f / 256.f) - mu * mu;
    sMV[r * 2 + 0] = mu;
    sMV[r * 2 + 1] = rsqrtf(var + 1e-5f);
  }
  __syncthreads();
  // normalize + gelu + store
  {
    int r = tid >> 3, seg = tid & 7;
    float mu = sMV[r * 2 + 0], inv = sMV[r * 2 + 1];
#pragma unroll 8
    for (int j = 0; j < 32; ++j) {
      int col = seg * 32 + j;
      float x = (sH[r * 256 + col] - mu) * inv * ln_g[(size_t)s * 256 + col] +
                ln_b[(size_t)s * 256 + col];
      float g = 0.5f * x * (1.f + erff(x * 0.70710678f));
      scales_out[((size_t)s * BB + b0 + r) * 256 + col] = g;
    }
  }
}

// ---------------------------------------------------------------------------
// K6: out[b] = LN( sum_s w[b,s]*scales[s,b,:] ; on_g, on_b )
// ---------------------------------------------------------------------------
__global__ __launch_bounds__(256) void k_out(const float* __restrict__ scales,
                                             const float* __restrict__ wts,
                                             const float* __restrict__ on_g,
                                             const float* __restrict__ on_b,
                                             float* __restrict__ out) {
  __shared__ float sV[256];
  int b = blockIdx.x, e = threadIdx.x;
  float w0 = wts[(size_t)b * 3 + 0];
  float w1 = wts[(size_t)b * 3 + 1];
  float w2 = wts[(size_t)b * 3 + 2];
  float v = w0 * scales[((size_t)0 * BB + b) * 256 + e] +
            w1 * scales[((size_t)1 * BB + b) * 256 + e] +
            w2 * scales[((size_t)2 * BB + b) * 256 + e];
  sV[e] = v;
  __syncthreads();
  for (int st = 128; st > 0; st >>= 1) {
    if (e < st) sV[e] += sV[e + st];
    __syncthreads();
  }
  float mu = sV[0] * (1.f / 256.f);
  __syncthreads();
  sV[e] = (v - mu) * (v - mu);
  __syncthreads();
  for (int st = 128; st > 0; st >>= 1) {
    if (e < st) sV[e] += sV[e + st];
    __syncthreads();
  }
  float var = sV[0] * (1.f / 256.f);
  out[(size_t)b * 256 + e] = (v - mu) * rsqrtf(var + 1e-5f) * on_g[e] + on_b[e];
}

// ---------------------------------------------------------------------------
// Host launcher
// ---------------------------------------------------------------------------
extern "C" void kernel_launch(void* const* d_in, const int* in_sizes, int n_in,
                              void* d_out, int out_size, void* d_ws, size_t ws_size,
                              hipStream_t stream) {
  const float* e_emb   = (const float*)d_in[0];
  const float* nb_rel  = (const float*)d_in[1];
  const float* delta_t = (const float*)d_in[2];
  const float* r_emb   = (const float*)d_in[3];
  const int*   h_mask  = (const int*)d_in[4];
  const float* rel_w   = (const float*)d_in[5];
  const float* lt1     = (const float*)d_in[6];
  const float* lt2     = (const float*)d_in[7];
  const float* shp     = (const float*)d_in[8];
  const float* lg      = (const float*)d_in[9];
  const float* attn_w  = (const float*)d_in[10];
  const float* sp_w    = (const float*)d_in[11];
  const float* sp_b    = (const float*)d_in[12];
  const float* ln_g    = (const float*)d_in[13];
  const float* ln_b    = (const float*)d_in[14];
  const float* csq_w   = (const float*)d_in[15];
  const float* csq_b   = (const float*)d_in[16];
  const float* csl_w   = (const float*)d_in[17];
  const float* csl_b   = (const float*)d_in[18];
  const float* on_g    = (const float*)d_in[19];
  const float* on_b    = (const float*)d_in[20];
  float* out = (float*)d_out;

  char* w = (char*)d_ws;
  unsigned short* ws_relw  = (unsigned short*)(w + 0);          // 128 KB
  unsigned short* ws_spw   = (unsigned short*)(w + 131072);     // 2.25 MB
  float* ws_qkey           = (float*)(w + 2490368);             // 1 MB
  float* ws_qproj          = (float*)(w + 3538944);             // 3 MB
  float* ws_wts            = (float*)(w + 6684672);             // 12 KB
  unsigned short* ws_feats = (unsigned short*)(w + 6696960);    // 9 MB
  float* ws_scales         = (float*)(w + 16134144);            // 3 MB

  hipLaunchKernelGGL(k_pack_relw, dim3(256), dim3(256), 0, stream, rel_w, ws_relw);
  hipLaunchKernelGGL(k_pack_spw, dim3(4608), dim3(256), 0, stream, sp_w, ws_spw);
  hipLaunchKernelGGL(k_qkey, dim3(BB), dim3(256), 0, stream, r_emb, rel_w, ws_qkey);
  hipLaunchKernelGGL(k_qproj, dim3(BB, 3), dim3(256), 0, stream, ws_qkey, attn_w, ws_qproj);
  hipLaunchKernelGGL(k_csw, dim3(BB), dim3(256), 0, stream, r_emb, csq_w, csq_b, csl_w,
                     csl_b, ws_wts);
  hipLaunchKernelGGL(k_main, dim3(BB), dim3(256), 0, stream, e_emb, nb_rel, delta_t,
                     h_mask, lt1, lt2, shp, lg, ws_relw, ws_qproj, ws_feats);
  hipLaunchKernelGGL(k_sp, dim3(BB / 32, 3), dim3(256), 0, stream, ws_feats, ws_spw,
                     sp_b, ln_g, ln_b, ws_scales);
  hipLaunchKernelGGL(k_out, dim3(BB), dim3(256), 0, stream, ws_scales, ws_wts, on_g,
                     on_b, out);
}
